// MHSA_9002251453132
// MI455X (gfx1250) — matrix operand order repaired and checked by reading, not compiled, not run on hardware
//
#include <hip/hip_runtime.h>
#include <hip/hip_bf16.h>
#include <stdint.h>

// ---------------------------------------------------------------------------
// CDNA5 (gfx1250) fused MHSA: bf16 WMMA projections + flash-style attention
// with double-buffered GLOBAL_LOAD_ASYNC_TO_LDS staging (ASYNCcnt pipeline).
// Wave32 only. Matrix op: v_wmma_f32_16x16x32_bf16.
// ---------------------------------------------------------------------------

typedef __attribute__((ext_vector_type(16))) __bf16 v16bf;
typedef __attribute__((ext_vector_type(8)))  float  v8f;

union V16BF { v16bf v; unsigned short us[16]; uint4 q[2]; };

__device__ __forceinline__ unsigned short f2bfu(float f) {
  union { float f; unsigned u; } x; x.f = f;
  unsigned u = x.u;
  unsigned r = u + 0x7FFFu + ((u >> 16) & 1u);   // round-to-nearest-even
  return (unsigned short)(r >> 16);
}

#define WMMA_BF16(C_, A_, B_) \
  __builtin_amdgcn_wmma_f32_16x16x32_bf16(false, (A_), false, (B_), (short)0, (C_), false, false)

// Async DMA: global -> LDS, 16 bytes, tracked by ASYNCcnt.
// VDST = wave-relative LDS byte address, GV addressing (64-bit vaddr).
__device__ __forceinline__ void async_b128(unsigned lds_byte_off, const void* g) {
  asm volatile("global_load_async_to_lds_b128 %0, %1, off"
               :: "v"(lds_byte_off), "v"(g) : "memory");
}
__device__ __forceinline__ void wait_async0() {
  asm volatile("s_wait_asynccnt 0" ::: "memory");
}
__device__ __forceinline__ unsigned lds_off(const void* p) {
  // Generic shared pointer: addr[63:32] = LDS aperture, addr[31:0] = LDS offset.
  return (unsigned)(size_t)p;
}

// Problem constants
#define BATCH 8
#define CH    512
#define HEADS 8
#define DHEAD 64
#define NPIX  1024   // 32*32

// ---------------------------------------------------------------------------
// Kernel 0: precision conversion + layout prep.
//   WQb/WKb/WVb : bf16 [512][512]           (row-major, A-operand friendly)
//   relT        : bf16 [h][N][d]            (rel_h+rel_w folded)
//   xT          : bf16 [b][N][C]            (channel-contiguous per pixel)
// ---------------------------------------------------------------------------
__global__ void mhsa_prep_kernel(const float* __restrict__ x,
                                 const float* __restrict__ Wq,
                                 const float* __restrict__ Wk,
                                 const float* __restrict__ Wv,
                                 const float* __restrict__ relh,
                                 const float* __restrict__ relw,
                                 unsigned short* __restrict__ xT,
                                 unsigned short* __restrict__ WQb,
                                 unsigned short* __restrict__ WKb,
                                 unsigned short* __restrict__ WVb,
                                 unsigned short* __restrict__ relT) {
  const int NW   = CH * CH;                 // 262144 per weight matrix
  const int NREL = HEADS * NPIX * DHEAD;    // 524288
  const int NXT  = BATCH * NPIX * CH;       // 4194304
  const int total = 3 * NW + NREL + NXT;
  for (int i = blockIdx.x * blockDim.x + threadIdx.x; i < total;
       i += gridDim.x * blockDim.x) {
    if (i < NW) {
      WQb[i] = f2bfu(Wq[i]);
    } else if (i < 2 * NW) {
      WKb[i - NW] = f2bfu(Wk[i - NW]);
    } else if (i < 3 * NW) {
      WVb[i - 2 * NW] = f2bfu(Wv[i - 2 * NW]);
    } else if (i < 3 * NW + NREL) {
      int idx = i - 3 * NW;
      int h  = idx >> 16;           // / (1024*64)
      int rm = idx & 65535;
      int n  = rm >> 6;
      int dd = rm & 63;
      int hh = n & 31;              // n = w*32 + hh
      int w  = n >> 5;
      relT[idx] = f2bfu(relh[(h * 64 + dd) * 32 + hh] +
                        relw[(h * 64 + dd) * 32 + w]);
    } else {
      int idx = i - 3 * NW - NREL;
      int c = idx & 511;
      int p = (idx >> 9) & 1023;
      int b = idx >> 19;
      xT[idx] = f2bfu(x[(size_t)(b * CH + c) * NPIX + p]);
    }
  }
}

// ---------------------------------------------------------------------------
// Kernel 1: fused QKV projection (1x1 conv == GEMM), bf16 WMMA.
//   D(o,p) = sum_c W(o,c) * xT(p,c) + bias(o)
// x-tile (16 pixels x 512 ch, 16KB) async-staged to LDS once per block and
// shared by all 4 waves; W A-operand loads software-pipelined one step ahead.
//   Q,K stored [b][h][N][d]   (d contiguous)
//   V   stored [b][h][d][N]   (N contiguous)
// ---------------------------------------------------------------------------
__global__ void __launch_bounds__(128) mhsa_qkv_proj_kernel(
    const unsigned short* __restrict__ xT,
    const unsigned short* __restrict__ WQb,
    const unsigned short* __restrict__ WKb,
    const unsigned short* __restrict__ WVb,
    const float* __restrict__ bq,
    const float* __restrict__ bk,
    const float* __restrict__ bv,
    unsigned short* __restrict__ Q,
    unsigned short* __restrict__ K,
    unsigned short* __restrict__ V) {
  __shared__ __align__(16) unsigned short xbuf[16][CH];   // 16 KB

  const int tid  = threadIdx.x;
  const int lane = tid & 31;
  const int wave = tid >> 5;
  const int ln   = lane & 15;
  const int half = lane >> 4;

  const int bid = blockIdx.x;
  const int b   = bid >> 9;          // / (64*8)
  const int rem = bid & 511;
  const int pt  = rem >> 3;          // pixel tile (16 pixels)
  const int h   = rem & 7;           // head == 64-channel output group
  const int o0  = h * 64 + wave * 16;

  // --- async stage x-tile: rows are 1KB contiguous; 1024 x 16B chunks -----
  {
    const unsigned short* xg = xT + (size_t)(b * NPIX + pt * 16) * CH;
    const unsigned xb = lds_off(&xbuf[0][0]);
#pragma unroll
    for (int c0 = 0; c0 < 1024; c0 += 128) {
      const int c = c0 + tid;
      async_b128(xb + c * 16, xg + (size_t)c * 8);
    }
  }

  const unsigned short* wqrow = WQb + (size_t)(o0 + ln) * CH;
  const unsigned short* wkrow = WKb + (size_t)(o0 + ln) * CH;
  const unsigned short* wvrow = WVb + (size_t)(o0 + ln) * CH;

  // Preload A operands for cs = 0 while the DMA runs.
  V16BF aqc, akc, avc, aqn, akn, avn;
  aqc.q[0] = *(const uint4*)(wqrow + half * 8);
  aqc.q[1] = *(const uint4*)(wqrow + 16 + half * 8);
  akc.q[0] = *(const uint4*)(wkrow + half * 8);
  akc.q[1] = *(const uint4*)(wkrow + 16 + half * 8);
  avc.q[0] = *(const uint4*)(wvrow + half * 8);
  avc.q[1] = *(const uint4*)(wvrow + 16 + half * 8);

  wait_async0();
  __syncthreads();

  v8f accq = {}; v8f acck = {}; v8f accv = {};

  for (int cs = 0; cs < CH / 32; ++cs) {
    const int c0 = cs * 32;
    if (cs + 1 < CH / 32) {           // pipeline next A operands
      const int c1 = c0 + 32;
      aqn.q[0] = *(const uint4*)(wqrow + c1 + half * 8);
      aqn.q[1] = *(const uint4*)(wqrow + c1 + 16 + half * 8);
      akn.q[0] = *(const uint4*)(wkrow + c1 + half * 8);
      akn.q[1] = *(const uint4*)(wkrow + c1 + 16 + half * 8);
      avn.q[0] = *(const uint4*)(wvrow + c1 + half * 8);
      avn.q[1] = *(const uint4*)(wvrow + c1 + 16 + half * 8);
    }
    // B operand from LDS: column = pixel ln, K(=c) runs of 8 at 8*half and 16+8*half.
    V16BF bx;
    bx.q[0] = *(const uint4*)(&xbuf[ln][c0 + half * 8]);
    bx.q[1] = *(const uint4*)(&xbuf[ln][c0 + 16 + half * 8]);

    accq = WMMA_BF16(accq, aqc.v, bx.v);
    acck = WMMA_BF16(acck, akc.v, bx.v);
    accv = WMMA_BF16(accv, avc.v, bx.v);

    aqc = aqn; akc = akn; avc = avn;
  }

  const int p = pt * 16 + ln;

  // Bias add: D-layout row M = r + 8*half.
#pragma unroll
  for (int r = 0; r < 8; ++r) {
    const int o = o0 + 8 * half + r;
    accq[r] += bq[o];
    acck[r] += bk[o];
    accv[r] += bv[o];
  }

  // Q/K: [b][h][p][dd], dd = wave*16 + 8*half + r -> 8 contiguous bf16.
  {
    union { unsigned short us[8]; uint4 q; } tq, tk;
#pragma unroll
    for (int r = 0; r < 8; ++r) { tq.us[r] = f2bfu(accq[r]); tk.us[r] = f2bfu(acck[r]); }
    const size_t base =
        (((size_t)(b * HEADS + h)) * NPIX + p) * DHEAD + wave * 16 + 8 * half;
    *(uint4*)(Q + base) = tq.q;
    *(uint4*)(K + base) = tk.q;
  }
  // V: [b][h][dd][p]
#pragma unroll
  for (int r = 0; r < 8; ++r) {
    const int dd = wave * 16 + 8 * half + r;
    V[(((size_t)(b * HEADS + h)) * DHEAD + dd) * NPIX + p] = f2bfu(accv[r]);
  }
}

// ---------------------------------------------------------------------------
// Kernel 2: flash attention. L[i,j] = q_i.k_j + rel_i.q_j ; P = softmax_j ;
// O = P * V^T. One wave = 16 query rows; 4 waves share one (b,h).
// Key-side chunks (K rows, Q rows as cp "keys", V columns) are async-DMA'd
// into double-buffered LDS and consumed by all 4 waves.
// grid = B * HEADS * (N/64); 4 waves/block.
// ---------------------------------------------------------------------------
__global__ void __launch_bounds__(128) mhsa_attn_kernel(
    const unsigned short* __restrict__ Q,
    const unsigned short* __restrict__ K,
    const unsigned short* __restrict__ V,
    const unsigned short* __restrict__ relT,
    float* __restrict__ out) {
  __shared__ __align__(16) unsigned short Kbuf[2][32][DHEAD];   // 2 x 4 KB
  __shared__ __align__(16) unsigned short Qkbuf[2][32][DHEAD];  // 2 x 4 KB
  __shared__ __align__(16) unsigned short Vbuf[2][DHEAD][32];   // 2 x 4 KB
  __shared__ __align__(16) unsigned short pbuf[4][16][32];      // P restage

  const int tid  = threadIdx.x;
  const int lane = tid & 31;
  const int wave = tid >> 5;
  const int ln   = lane & 15;
  const int half = lane >> 4;

  const int bid = blockIdx.x;
  const int b   = bid >> 7;          // / (8*16)
  const int rem = bid & 127;
  const int h   = rem >> 4;
  const int mb  = rem & 15;
  const int m0  = mb * 64 + wave * 16;
  const size_t bh = (size_t)(b * HEADS + h);

  const unsigned short* Kg = K + bh * NPIX * DHEAD;
  const unsigned short* Qg = Q + bh * NPIX * DHEAD;
  const unsigned short* Vg = V + bh * DHEAD * NPIX;
  const unsigned kb = lds_off(&Kbuf[0][0][0]);
  const unsigned qb = lds_off(&Qkbuf[0][0][0]);
  const unsigned vb = lds_off(&Vbuf[0][0][0]);

  // 32-key chunk = 4KB per tensor = 256 x 16B chunks; 6 async issues/thread.
  auto issue_chunk = [&](int j0, int buf) {
#pragma unroll
    for (int c0 = 0; c0 < 256; c0 += 128) {
      const int c = c0 + tid;
      async_b128(kb + buf * 4096 + c * 16, Kg + (size_t)j0 * DHEAD + c * 8);
      async_b128(qb + buf * 4096 + c * 16, Qg + (size_t)j0 * DHEAD + c * 8);
      async_b128(vb + buf * 4096 + c * 16,
                 Vg + (size_t)(c >> 2) * NPIX + j0 + (c & 3) * 8);
    }
  };

  // Resident A operands: Q rows and Rel rows for this wave's 16 queries.
  const unsigned short* qrow = Q    + (bh * NPIX + m0 + ln) * DHEAD;
  const unsigned short* rrow = relT + ((size_t)h * NPIX + m0 + ln) * DHEAD;

  issue_chunk(0, 0);                 // prefetch first chunk

  V16BF aq[2], ar[2];
#pragma unroll
  for (int s = 0; s < 2; ++s) {
    aq[s].q[0] = *(const uint4*)(qrow + s * 32 + half * 8);
    aq[s].q[1] = *(const uint4*)(qrow + s * 32 + 16 + half * 8);
    ar[s].q[0] = *(const uint4*)(rrow + s * 32 + half * 8);
    ar[s].q[1] = *(const uint4*)(rrow + s * 32 + 16 + half * 8);
  }

  v8f o_acc[4] = {};                 // 16 rows x 64 d-columns
  float rowm[8], rowl[8];
#pragma unroll
  for (int r = 0; r < 8; ++r) { rowm[r] = -3.0e38f; rowl[r] = 0.0f; }

  for (int i = 0; i < NPIX / 32; ++i) {
    const int cur = i & 1;
    const int j0  = i * 32;

    wait_async0();                   // chunk i landed (wave-local)
    __syncthreads();                 // visible block-wide; prev reads done
    if (i + 1 < NPIX / 32) issue_chunk(j0 + 32, 1 - cur);  // overlap DMA

    // --- logits: 16 x 32 strip, two 16x16 tiles (B operands from LDS) ----
    v8f sacc[2] = {};
#pragma unroll
    for (int jt = 0; jt < 2; ++jt) {
      const int jr = jt * 16 + ln;   // key row within chunk
#pragma unroll
      for (int s = 0; s < 2; ++s) {
        V16BF bk_, bq_;
        bk_.q[0] = *(const uint4*)(&Kbuf[cur][jr][s * 32 + half * 8]);
        bk_.q[1] = *(const uint4*)(&Kbuf[cur][jr][s * 32 + 16 + half * 8]);
        bq_.q[0] = *(const uint4*)(&Qkbuf[cur][jr][s * 32 + half * 8]);
        bq_.q[1] = *(const uint4*)(&Qkbuf[cur][jr][s * 32 + 16 + half * 8]);
        sacc[jt] = WMMA_BF16(sacc[jt], aq[s].v, bk_.v);
        sacc[jt] = WMMA_BF16(sacc[jt], ar[s].v, bq_.v);
      }
    }

    // --- online softmax (rows live across 16 lanes of a half-wave) -------
    float scl[8], p0[8], p1[8];
#pragma unroll
    for (int r = 0; r < 8; ++r) {
      float t = fmaxf(sacc[0][r], sacc[1][r]);
      t = fmaxf(t, __shfl_xor(t, 1));
      t = fmaxf(t, __shfl_xor(t, 2));
      t = fmaxf(t, __shfl_xor(t, 4));
      t = fmaxf(t, __shfl_xor(t, 8));
      const float mn = fmaxf(rowm[r], t);
      scl[r] = __expf(rowm[r] - mn);
      rowm[r] = mn;
      p0[r] = __expf(sacc[0][r] - mn);
      p1[r] = __expf(sacc[1][r] - mn);
      float rs = p0[r] + p1[r];
      rs += __shfl_xor(rs, 1);
      rs += __shfl_xor(rs, 2);
      rs += __shfl_xor(rs, 4);
      rs += __shfl_xor(rs, 8);
      rowl[r] = rowl[r] * scl[r] + rs;
    }
#pragma unroll
    for (int t = 0; t < 4; ++t)
#pragma unroll
      for (int r = 0; r < 8; ++r) o_acc[t][r] *= scl[r];

    // --- restage P (D-layout f32) -> A-layout bf16 via LDS ---------------
#pragma unroll
    for (int r = 0; r < 8; ++r) {
      pbuf[wave][8 * half + r][ln]      = f2bfu(p0[r]);
      pbuf[wave][8 * half + r][16 + ln] = f2bfu(p1[r]);
    }
    asm volatile("s_wait_dscnt 0" ::: "memory");
    V16BF ap;
    ap.q[0] = *(const uint4*)(&pbuf[wave][ln][half * 8]);
    ap.q[1] = *(const uint4*)(&pbuf[wave][ln][16 + half * 8]);

    // --- O += P (16x32) x V^T (32n x 16dd), V from LDS -------------------
#pragma unroll
    for (int t = 0; t < 4; ++t) {
      V16BF bv_;
      bv_.q[0] = *(const uint4*)(&Vbuf[cur][t * 16 + ln][half * 8]);
      bv_.q[1] = *(const uint4*)(&Vbuf[cur][t * 16 + ln][16 + half * 8]);
      o_acc[t] = WMMA_BF16(o_acc[t], ap.v, bv_.v);
    }
  }

  // --- finalize: out[b, h*64+dd, m] = O[m,dd] / rowl[m] -------------------
#pragma unroll
  for (int t = 0; t < 4; ++t) {
    const int dd = t * 16 + ln;
    float o_[8];
#pragma unroll
    for (int r = 0; r < 8; ++r) o_[r] = o_acc[t][r] / rowl[r];
    float* obase = out + ((size_t)(b * CH + h * 64 + dd)) * NPIX + m0 + 8 * half;
    *(float4*)(obase)     = make_float4(o_[0], o_[1], o_[2], o_[3]);
    *(float4*)(obase + 4) = make_float4(o_[4], o_[5], o_[6], o_[7]);
  }
}

// ---------------------------------------------------------------------------
extern "C" void kernel_launch(void* const* d_in, const int* in_sizes, int n_in,
                              void* d_out, int out_size, void* d_ws, size_t ws_size,
                              hipStream_t stream) {
  const float* x    = (const float*)d_in[0];
  const float* Wq   = (const float*)d_in[1];
  const float* bq   = (const float*)d_in[2];
  const float* Wk   = (const float*)d_in[3];
  const float* bk   = (const float*)d_in[4];
  const float* Wv   = (const float*)d_in[5];
  const float* bv   = (const float*)d_in[6];
  const float* relh = (const float*)d_in[7];
  const float* relw = (const float*)d_in[8];
  float* out = (float*)d_out;

  // Workspace layout (bytes):
  //   xT   bf16 [8][1024][512] : 8,388,608
  //   WQb  bf16 [512][512]     :   524,288   (x3)
  //   relT bf16 [8][1024][64]  : 1,048,576
  //   Q,K  bf16 [8][8][1024][64], V bf16 [8][8][64][1024] : 8,388,608 each
  unsigned char* ws = (unsigned char*)d_ws;
  unsigned short* xT   = (unsigned short*)(ws);
  unsigned short* WQb  = (unsigned short*)(ws + 8388608u);
  unsigned short* WKb  = (unsigned short*)(ws + 8912896u);
  unsigned short* WVb  = (unsigned short*)(ws + 9437184u);
  unsigned short* relT = (unsigned short*)(ws + 9961472u);
  unsigned short* Q    = (unsigned short*)(ws + 11010048u);
  unsigned short* K    = (unsigned short*)(ws + 19398656u);
  unsigned short* V    = (unsigned short*)(ws + 27787264u);
  (void)in_sizes; (void)n_in; (void)out_size; (void)ws_size;

  mhsa_prep_kernel<<<2048, 256, 0, stream>>>(x, Wq, Wk, Wv, relh, relw,
                                             xT, WQb, WKb, WVb, relT);
  mhsa_qkv_proj_kernel<<<4096, 128, 0, stream>>>(xT, WQb, WKb, WVb,
                                                 bq, bk, bv, Q, K, V);
  mhsa_attn_kernel<<<1024, 128, 0, stream>>>(Q, K, V, relT, out);
}
